// Attention_8521215115924
// MI455X (gfx1250) — matrix operand order repaired and checked by reading, not compiled, not run on hardware
//
#include <hip/hip_runtime.h>

// ---------------------------------------------------------------------------
// Attention block for MI455X (gfx1250, wave32, WMMA).
// GEMMs use v_wmma_f32_16x16x32_bf16 (bf16 A/B, f32 accum).
// The two large GEMMs (energy = QK^T, ctx = aV) stage their B operand through
// LDS with global_load_async_to_lds_b128 (ASYNCcnt) double buffering.
// ---------------------------------------------------------------------------

typedef __attribute__((ext_vector_type(16))) __bf16 v16bf;
typedef __attribute__((ext_vector_type(8)))  __bf16 v8bf;
typedef __attribute__((ext_vector_type(8)))  float  v8f;
typedef __attribute__((ext_vector_type(4)))  float  v4f;

namespace {
constexpr int BB = 8;      // batch
constexpr int HH = 512;    // hid
constexpr int EE = 256;    // emb
constexpr int TT = 2048;   // dec time
constexpr int SS = 2048;   // enc time
constexpr float SCALE = 0.8366600265340756f; // sqrt(0.7)

constexpr int CK     = 64;  // K chunk staged per double-buffer step
constexpr int BPITCH = 72;  // LDS row pitch (bf16 elems): 144B -> 16B aligned,
                            // 36-dword bank stride -> conflict-free 16-row reads
}

#define WMMA_BF16(A_, B_, C_) \
  __builtin_amdgcn_wmma_f32_16x16x32_bf16(false, (A_), false, (B_), (short)0, (C_), false, false)

// --- CDNA5 async global->LDS copy (ASYNCcnt) -------------------------------
__device__ __forceinline__ void async_ld_b128(unsigned lds_addr, const void* gaddr) {
  asm volatile("global_load_async_to_lds_b128 %0, %1, off"
               :: "v"(lds_addr), "v"((unsigned long long)(uintptr_t)gaddr)
               : "memory");
}
template <int N>
__device__ __forceinline__ void wait_asynccnt() {
  asm volatile("s_wait_asynccnt %0" :: "n"(N) : "memory");
}

// --- fragment loaders (contiguous k order, k = 16*(lane/16) + s, same for A and B) -----------

// A 16x32 bf16, row-major [M,K] source. lane<16: rows, K = kh*16..+15
__device__ __forceinline__ v16bf load_a_bf16(const __bf16* p, int lda, int kk, int lane) {
  const int m  = lane & 15;
  const int kh = lane >> 4;
  const __bf16* q = p + (size_t)m * lda + kk + kh * 16;
  v8bf lo = *(const v8bf*)q;          // 16B
  v8bf hi = *(const v8bf*)(q + 8);    // 16B
  v16bf r;
#pragma unroll
  for (int i = 0; i < 8; ++i) { r[i] = lo[i]; r[i + 8] = hi[i]; }
  return r;
}

// A 16x32 from fp32 row-major source, converted to bf16 on the fly.
__device__ __forceinline__ v16bf load_a_f32(const float* p, int lda, int kk, int lane) {
  const int m  = lane & 15;
  const int kh = lane >> 4;
  const float* q = p + (size_t)m * lda + kk + kh * 16;
  v4f x0 = *(const v4f*)q;
  v4f x1 = *(const v4f*)(q + 4);
  v4f x2 = *(const v4f*)(q + 8);
  v4f x3 = *(const v4f*)(q + 12);
  v16bf r;
#pragma unroll
  for (int i = 0; i < 4; ++i) {
    r[i]      = (__bf16)x0[i];
    r[i + 4]  = (__bf16)x1[i];
    r[i + 8]  = (__bf16)x2[i];
    r[i + 12] = (__bf16)x3[i];
  }
  return r;
}

// B 32x16 bf16: source row-major [N, K] (N rows, contiguous K). lane n = column,
// lane<16 holds K = 0..15, lane>=16 holds K = 16..31 -> one 32B load per lane.
__device__ __forceinline__ v16bf load_b_bf16(const __bf16* p, int ldb, int kk, int lane) {
  const int n  = lane & 15;
  const int kh = lane >> 4;
  return *(const v16bf*)(p + (size_t)n * ldb + kk + kh * 16);
}

// Same fragment shape, sourced from an LDS-staged chunk (pitch BPITCH).
__device__ __forceinline__ v16bf load_b_lds(const __bf16* p, int kk, int lane) {
  const int n  = lane & 15;
  const int kh = lane >> 4;
  return *(const v16bf*)(p + n * BPITCH + kk + kh * 16);
}

// Stage a [128 x CK] bf16 B chunk into LDS (whole 256-thread block, 4 async
// b128 copies per thread). src = tile origin, ld = source row stride (elems).
__device__ __forceinline__ void stage_b_chunk(const __bf16* src, int ld, int kk,
                                              __bf16* buf, int tid) {
#pragma unroll
  for (int j = 0; j < 4; ++j) {
    const int idx = j * 256 + tid;  // 0..1023 -> (row, 16B segment)
    const int row = idx >> 3;
    const int seg = idx & 7;
    const __bf16* g = src + (size_t)row * ld + kk + seg * 8;
    const unsigned l = (unsigned)(uintptr_t)(buf + row * BPITCH + seg * 8);
    async_ld_b128(l, g);
  }
}

// --- data movement prep ----------------------------------------------------

// in: [R, C] fp32 row-major  ->  out: [C, R] bf16  (LDS-tiled transpose)
__global__ __launch_bounds__(256)
void transpose_to_bf16(const float* __restrict__ in, __bf16* __restrict__ out,
                       int R, int C) {
  __shared__ float tile[32][33];
  const int b  = blockIdx.z;
  const int c0 = blockIdx.x * 32;
  const int r0 = blockIdx.y * 32;
  in  += (size_t)b * R * C;
  out += (size_t)b * R * C;
  const int tx = threadIdx.x, ty = threadIdx.y; // 32 x 8
#pragma unroll
  for (int i = 0; i < 32; i += 8)
    tile[ty + i][tx] = in[(size_t)(r0 + ty + i) * C + c0 + tx];
  __syncthreads();
#pragma unroll
  for (int i = 0; i < 32; i += 8)
    out[(size_t)(c0 + ty + i) * R + r0 + tx] = (__bf16)tile[tx][ty + i];
}

__global__ __launch_bounds__(256)
void cvt_f32_to_bf16(const float* __restrict__ in, __bf16* __restrict__ out, size_t n) {
  size_t i = (size_t)blockIdx.x * blockDim.x + threadIdx.x;
  size_t stride = (size_t)gridDim.x * blockDim.x;
  for (; i < n; i += stride) out[i] = (__bf16)in[i];
}

// --- GEMM 1: Q = (dcT @ Wh2e^T + b_h2e + embedd) * SCALE -> bf16 [B,T,E] ---
__global__ __launch_bounds__(256)
void q_gemm(const __bf16* __restrict__ dcT,   // [B,T,H]
            const __bf16* __restrict__ Wh2e,  // [E,H]
            const float*  __restrict__ bh2e,  // [E]
            const float*  __restrict__ embedd,// [B,T,E]
            __bf16* __restrict__ Qbf) {       // [B,T,E]
  const int b  = blockIdx.z;
  const int wid = threadIdx.x >> 5, lane = threadIdx.x & 31;
  const int tw = blockIdx.x * 64 + (wid & 1) * 32;
  const int ew = blockIdx.y * 128 + (wid >> 1) * 32;
  const __bf16* A  = dcT + ((size_t)b * TT + tw) * HH;
  const __bf16* Bm = Wh2e + (size_t)ew * HH;
  v8f acc[2][2] = {};
  for (int kk = 0; kk < HH; kk += 32) {
    v16bf a0 = load_a_bf16(A, HH, kk, lane);
    v16bf a1 = load_a_bf16(A + (size_t)16 * HH, HH, kk, lane);
    v16bf b0 = load_b_bf16(Bm, HH, kk, lane);
    v16bf b1 = load_b_bf16(Bm + (size_t)16 * HH, HH, kk, lane);
    acc[0][0] = WMMA_BF16(a0, b0, acc[0][0]);
    acc[0][1] = WMMA_BF16(a0, b1, acc[0][1]);
    acc[1][0] = WMMA_BF16(a1, b0, acc[1][0]);
    acc[1][1] = WMMA_BF16(a1, b1, acc[1][1]);
  }
  const int mrow = (lane >> 4) * 8, ncol = lane & 15;
#pragma unroll
  for (int mt = 0; mt < 2; ++mt)
#pragma unroll
    for (int nt = 0; nt < 2; ++nt)
#pragma unroll
      for (int r = 0; r < 8; ++r) {
        const int t = tw + mt * 16 + mrow + r;
        const int e = ew + nt * 16 + ncol;
        const size_t idx = ((size_t)b * TT + t) * EE + e;
        Qbf[idx] = (__bf16)((acc[mt][nt][r] + bh2e[e] + embedd[idx]) * SCALE);
      }
}

// --- GEMM 2: energy[b,t,s] = sum_e Q[b,t,e]*K[b,s,e] -> fp32 (pre-softmax)
// B operand (K tiles) async-staged through LDS, double buffered.
__global__ __launch_bounds__(256)
void energy_gemm(const __bf16* __restrict__ Qbf,  // [B,T,E]
                 const __bf16* __restrict__ Kbf,  // [B,S,E]
                 float* __restrict__ aout) {      // [B,T,S]
  __shared__ __bf16 bbuf[2][128 * BPITCH];        // 2 x 18KB
  const int b   = blockIdx.z;
  const int tid = threadIdx.x;
  const int wid = tid >> 5, lane = tid & 31;
  const int tw = blockIdx.x * 64 + (wid & 1) * 32;
  const int s0 = blockIdx.y * 128;
  const __bf16* A    = Qbf + ((size_t)b * TT + tw) * EE;
  const __bf16* Bsrc = Kbf + ((size_t)b * SS + s0) * EE;
  const int nwin = (wid >> 1) * 32;  // this wave's N window in the 128 tile

  v8f acc[2][2] = {};
  constexpr int NCH = EE / CK;  // 4 chunks
  stage_b_chunk(Bsrc, EE, 0, bbuf[0], tid);
  for (int c = 0; c < NCH; ++c) {
    if (c + 1 < NCH) {
      stage_b_chunk(Bsrc, EE, (c + 1) * CK, bbuf[(c + 1) & 1], tid);
      wait_asynccnt<4>();   // in-order retire: chunk c complete
    } else {
      wait_asynccnt<0>();
    }
    __syncthreads();
    const __bf16* bw = bbuf[c & 1] + nwin * BPITCH;
#pragma unroll
    for (int kk = 0; kk < CK; kk += 32) {
      v16bf a0 = load_a_bf16(A, EE, c * CK + kk, lane);
      v16bf a1 = load_a_bf16(A + (size_t)16 * EE, EE, c * CK + kk, lane);
      v16bf b0 = load_b_lds(bw, kk, lane);
      v16bf b1 = load_b_lds(bw + 16 * BPITCH, kk, lane);
      acc[0][0] = WMMA_BF16(a0, b0, acc[0][0]);
      acc[0][1] = WMMA_BF16(a0, b1, acc[0][1]);
      acc[1][0] = WMMA_BF16(a1, b0, acc[1][0]);
      acc[1][1] = WMMA_BF16(a1, b1, acc[1][1]);
    }
    __syncthreads();  // buffer (c&1) is re-staged next iteration
  }
  const int mrow = (lane >> 4) * 8, ncol = lane & 15;
#pragma unroll
  for (int mt = 0; mt < 2; ++mt)
#pragma unroll
    for (int nt = 0; nt < 2; ++nt)
#pragma unroll
      for (int r = 0; r < 8; ++r) {
        const int t = tw + mt * 16 + mrow + r;
        const int s = s0 + nwin + nt * 16 + ncol;
        aout[((size_t)b * TT + t) * SS + s] = acc[mt][nt][r];
      }
}

// --- softmax over last axis, in place (one block per row) ------------------
__global__ __launch_bounds__(256)
void softmax_rows(float* __restrict__ a) {
  __shared__ float red[256];
  float* p = a + (size_t)blockIdx.x * SS;
  const int tid = threadIdx.x;
  float m = -INFINITY;
  for (int i = tid; i < SS; i += 256) m = fmaxf(m, p[i]);
  red[tid] = m; __syncthreads();
  for (int s = 128; s > 0; s >>= 1) {
    if (tid < s) red[tid] = fmaxf(red[tid], red[tid + s]);
    __syncthreads();
  }
  m = red[0]; __syncthreads();
  float sum = 0.f;
  for (int i = tid; i < SS; i += 256) {
    float e = __expf(p[i] - m);
    p[i] = e;
    sum += e;
  }
  red[tid] = sum; __syncthreads();
  for (int s = 128; s > 0; s >>= 1) {
    if (tid < s) red[tid] += red[tid + s];
    __syncthreads();
  }
  const float inv = 1.0f / red[0];
  for (int i = tid; i < SS; i += 256) p[i] *= inv;
}

// --- GEMM 3: ctx[b,t,e] = sum_s a[b,t,s]*V[b,s,e] -> bf16 [B,T,E] ----------
// B operand (Vt tiles) async-staged through LDS, double buffered; K=2048.
__global__ __launch_bounds__(256)
void pv_gemm(const float*  __restrict__ a,    // [B,T,S] (softmaxed)
             const __bf16* __restrict__ Vt,   // [B,E,S]  (V transposed)
             __bf16* __restrict__ ctx) {      // [B,T,E]
  __shared__ __bf16 bbuf[2][128 * BPITCH];
  const int b   = blockIdx.z;
  const int tid = threadIdx.x;
  const int wid = tid >> 5, lane = tid & 31;
  const int tw = blockIdx.x * 64 + (wid & 1) * 32;
  const int e0 = blockIdx.y * 128;
  const float*  A    = a  + ((size_t)b * TT + tw) * SS;
  const __bf16* Bsrc = Vt + ((size_t)b * EE + e0) * SS;
  const int nwin = (wid >> 1) * 32;

  v8f acc[2][2] = {};
  constexpr int NCH = SS / CK;  // 32 chunks
  stage_b_chunk(Bsrc, SS, 0, bbuf[0], tid);
  for (int c = 0; c < NCH; ++c) {
    if (c + 1 < NCH) {
      stage_b_chunk(Bsrc, SS, (c + 1) * CK, bbuf[(c + 1) & 1], tid);
      wait_asynccnt<4>();
    } else {
      wait_asynccnt<0>();
    }
    __syncthreads();
    const __bf16* bw = bbuf[c & 1] + nwin * BPITCH;
#pragma unroll
    for (int kk = 0; kk < CK; kk += 32) {
      v16bf a0 = load_a_f32(A, SS, c * CK + kk, lane);
      v16bf a1 = load_a_f32(A + (size_t)16 * SS, SS, c * CK + kk, lane);
      v16bf b0 = load_b_lds(bw, kk, lane);
      v16bf b1 = load_b_lds(bw + 16 * BPITCH, kk, lane);
      acc[0][0] = WMMA_BF16(a0, b0, acc[0][0]);
      acc[0][1] = WMMA_BF16(a0, b1, acc[0][1]);
      acc[1][0] = WMMA_BF16(a1, b0, acc[1][0]);
      acc[1][1] = WMMA_BF16(a1, b1, acc[1][1]);
    }
    __syncthreads();
  }
  const int mrow = (lane >> 4) * 8, ncol = lane & 15;
#pragma unroll
  for (int mt = 0; mt < 2; ++mt)
#pragma unroll
    for (int nt = 0; nt < 2; ++nt)
#pragma unroll
      for (int r = 0; r < 8; ++r) {
        const int t = tw + mt * 16 + mrow + r;
        const int e = e0 + nwin + nt * 16 + ncol;
        ctx[((size_t)b * TT + t) * EE + e] = (__bf16)acc[mt][nt][r];
      }
}

// --- GEMM 4: out[b,h,t] = SCALE*(sum_e ctx[b,t,e]*We2h[h,e] + b_e2h[h] + dec[b,h,t])
// D tile is [t,h]; stage via LDS so global stores are contiguous along t.
__global__ __launch_bounds__(256)
void e2h_gemm(const __bf16* __restrict__ ctx,   // [B,T,E]
              const __bf16* __restrict__ We2h,  // [H,E]
              const float*  __restrict__ be2h,  // [H]
              const float*  __restrict__ dec,   // [B,H,T]
              float* __restrict__ outc) {       // [B,H,T]
  __shared__ float ldsbuf[8][32 * 33];
  const int b  = blockIdx.z;
  const int wid = threadIdx.x >> 5, lane = threadIdx.x & 31;
  const int tw = blockIdx.x * 64 + (wid & 1) * 32;
  const int hw = blockIdx.y * 128 + (wid >> 1) * 32;
  const __bf16* A  = ctx + ((size_t)b * TT + tw) * EE;
  const __bf16* Bm = We2h + (size_t)hw * EE;
  v8f acc[2][2] = {};
  for (int kk = 0; kk < EE; kk += 32) {
    v16bf a0 = load_a_bf16(A, EE, kk, lane);
    v16bf a1 = load_a_bf16(A + (size_t)16 * EE, EE, kk, lane);
    v16bf b0 = load_b_bf16(Bm, EE, kk, lane);
    v16bf b1 = load_b_bf16(Bm + (size_t)16 * EE, EE, kk, lane);
    acc[0][0] = WMMA_BF16(a0, b0, acc[0][0]);
    acc[0][1] = WMMA_BF16(a0, b1, acc[0][1]);
    acc[1][0] = WMMA_BF16(a1, b0, acc[1][0]);
    acc[1][1] = WMMA_BF16(a1, b1, acc[1][1]);
  }
  // wave-private LDS transpose: [t_local, h_local] -> [h_local][t_local]
  float* ldsw = &ldsbuf[wid][0];
  const int mrow = (lane >> 4) * 8, ncol = lane & 15;
#pragma unroll
  for (int mt = 0; mt < 2; ++mt)
#pragma unroll
    for (int nt = 0; nt < 2; ++nt)
#pragma unroll
      for (int r = 0; r < 8; ++r)
        ldsw[(nt * 16 + ncol) * 33 + mt * 16 + mrow + r] = acc[mt][nt][r];
  // LDS ops from the same wave are in order (DScnt); no barrier needed.
  const float* decb = dec  + (size_t)b * HH * TT;
  float*       outb = outc + (size_t)b * HH * TT;
#pragma unroll 4
  for (int i = 0; i < 32; ++i) {
    const int h = hw + i;
    const int t = tw + lane;
    const float v = ldsw[i * 33 + lane];
    outb[(size_t)h * TT + t] = (v + be2h[h] + decb[(size_t)h * TT + t]) * SCALE;
  }
}

// ---------------------------------------------------------------------------
extern "C" void kernel_launch(void* const* d_in, const int* in_sizes, int n_in,
                              void* d_out, int out_size, void* d_ws, size_t ws_size,
                              hipStream_t stream) {
  const float* dec_conved  = (const float*)d_in[0]; // [B,H,T]
  const float* embedd      = (const float*)d_in[1]; // [B,T,E]
  const float* en_conved   = (const float*)d_in[2]; // [B,S,E]
  const float* en_combined = (const float*)d_in[3]; // [B,S,E]
  const float* W_h2e       = (const float*)d_in[4]; // [E,H]
  const float* b_h2e       = (const float*)d_in[5]; // [E]
  const float* W_e2h       = (const float*)d_in[6]; // [H,E]
  const float* b_e2h       = (const float*)d_in[7]; // [H]

  float* outA = (float*)d_out;                          // a: [B,T,S]
  float* outC = outA + (size_t)BB * TT * SS;            // conved: [B,H,T]

  char* ws = (char*)d_ws;
  auto take = [&](size_t bytes) {
    char* p = ws;
    ws += (bytes + 255) & ~(size_t)255;
    return p;
  };
  __bf16* dcT   = (__bf16*)take((size_t)BB * TT * HH * 2); // [B,T,H]
  __bf16* Kbf   = (__bf16*)take((size_t)BB * SS * EE * 2); // [B,S,E]
  __bf16* Vt    = (__bf16*)take((size_t)BB * EE * SS * 2); // [B,E,S]
  __bf16* Wh2eb = (__bf16*)take((size_t)EE * HH * 2);      // [E,H]
  __bf16* We2hb = (__bf16*)take((size_t)HH * EE * 2);      // [H,E]
  __bf16* Qbf   = (__bf16*)take((size_t)BB * TT * EE * 2); // [B,T,E]
  __bf16* Ctx   = (__bf16*)take((size_t)BB * TT * EE * 2); // [B,T,E]
  (void)ws_size; (void)in_sizes; (void)n_in; (void)out_size;

  // prep: transposes + bf16 conversions
  transpose_to_bf16<<<dim3(TT / 32, HH / 32, BB), dim3(32, 8), 0, stream>>>(
      dec_conved, dcT, HH, TT);
  transpose_to_bf16<<<dim3(EE / 32, SS / 32, BB), dim3(32, 8), 0, stream>>>(
      en_combined, Vt, SS, EE);
  cvt_f32_to_bf16<<<1024, 256, 0, stream>>>(en_conved, Kbf, (size_t)BB * SS * EE);
  cvt_f32_to_bf16<<<128, 256, 0, stream>>>(W_h2e, Wh2eb, (size_t)EE * HH);
  cvt_f32_to_bf16<<<128, 256, 0, stream>>>(W_e2h, We2hb, (size_t)HH * EE);

  // 1) Q projection (+bias, +embedd, *scale)
  q_gemm<<<dim3(TT / 64, EE / 128, BB), 256, 0, stream>>>(
      dcT, Wh2eb, b_h2e, embedd, Qbf);
  // 2) energy = Q K^T (async-LDS staged B)
  energy_gemm<<<dim3(TT / 64, SS / 128, BB), 256, 0, stream>>>(Qbf, Kbf, outA);
  // 3) softmax rows (in place in d_out)
  softmax_rows<<<BB * TT, 256, 0, stream>>>(outA);
  // 4) ctx = a V (async-LDS staged B)
  pv_gemm<<<dim3(TT / 64, EE / 128, BB), 256, 0, stream>>>(outA, Vt, Ctx);
  // 5) out = scale*(ctx We2h^T + bias + dc), stored transposed [B,H,T]
  e2h_gemm<<<dim3(TT / 64, HH / 128, BB), 256, 0, stream>>>(
      Ctx, We2hb, b_e2h, dec_conved, outC);
}
